// Non_Max_Suppression_16106127360133
// MI455X (gfx1250) — 8-bit matrix operand order repaired and checked by reading, compiled, not run on hardware
//
#include <hip/hip_runtime.h>
#include <hip/hip_bf16.h>
#include <stdint.h>


#define N_BOXES 1024
#define BATCH   64
#define N_OBJ   20
#define OVERLAP_THR 0.3f
#define SCORE_THR   0.1f

typedef __attribute__((ext_vector_type(8))) int v8i;

__global__ __launch_bounds__(256) void build_mask_kernel(
    const float* __restrict__ bx, const float* __restrict__ by,
    const float* __restrict__ bw, const float* __restrict__ bh,
    uint8_t* __restrict__ mask)
{
    __shared__ float sx1[N_BOXES], sy1[N_BOXES], sx3[N_BOXES], sy3[N_BOXES], sar[N_BOXES];
    const int b = blockIdx.x;
    const int t = threadIdx.x;

    for (int i = t; i < N_BOXES; i += 256) {
        const int idx = i * BATCH + b;
        const float x = bx[idx], y = by[idx], w = bw[idx], h = bh[idx];
        sx1[i] = x - 0.5f * w; sx3[i] = x + 0.5f * w;
        sy1[i] = y - 0.5f * h; sy3[i] = y + 0.5f * h;
        sar[i] = w * h;
    }
    __syncthreads();

    const int i      = blockIdx.y * 128 + (t >> 1);
    const int jstart = (t & 1) * 512;
    const float x1 = sx1[i], x3 = sx3[i], y1 = sy1[i], y3 = sy3[i], ar = sar[i];
    uint8_t* row = mask + (((size_t)b << 10) + (size_t)i << 10);

    for (int j = jstart; j < jstart + 512; j += 4) {
        unsigned int pack = 0u;
#pragma unroll
        for (int e = 0; e < 4; ++e) {
            const int jj = j + e;
            const float ix = fminf(x3, sx3[jj]) - fmaxf(x1, sx1[jj]);
            const float iy = fminf(y3, sy3[jj]) - fmaxf(y1, sy1[jj]);
            const float inter = fmaxf(ix, 0.f) * fmaxf(iy, 0.f);
            const float mina  = fminf(ar, sar[jj]);
            if (inter > OVERLAP_THR * mina) pack |= (1u << (e * 8));
        }
        *(unsigned int*)(row + j) = pack;
    }
}

__global__ __launch_bounds__(256) void nms_select_kernel(
    const float* __restrict__ prob, const uint8_t* __restrict__ mask,
    float* __restrict__ out_mask)
{
    __shared__ float sScore[N_BOXES];
    __shared__ float sPoss[N_BOXES];
    __shared__ float sSp[N_BOXES];
    __shared__ float sSel[N_BOXES];
    __shared__ unsigned int sSelW[N_BOXES / 4];
    __shared__ int   sBlocks[N_BOXES];

    const int b    = blockIdx.x;
    const int t    = threadIdx.x;
    const int lane = t & 31;
    const int wave = t >> 5;
    const int half = (lane >= 16) ? 1 : 0;
    const uint8_t* bmask = mask + ((size_t)b << 20);

    for (int q = 0; q < 4; ++q) {
        const int i = t + q * 256;
        const float s = fmaxf(prob[i * BATCH + b], 0.f);
        sScore[i] = s;
        sPoss[i]  = (s > SCORE_THR) ? 1.f : 0.f;
        sSel[i]   = 0.f;
    }
    __syncthreads();

    for (int it = 0; it < N_OBJ; ++it) {
        for (int q = 0; q < 4; ++q) {
            const int j = t + q * 256;
            sSp[j] = sScore[j] * sPoss[j];
        }
        __syncthreads();

        for (int q = 0; q < 4; ++q) {
            const int i = t + q * 256;
            if (sPoss[i] != 0.f) {
                const uint4* row = (const uint4*)(bmask + ((size_t)i << 10));
                float best = -1.f; int bi = 0;
                for (int c = 0; c < 64; ++c) {
                    __builtin_prefetch((const char*)(row + c + 8), 0, 0);
                    const uint4 w = row[c];
                    if ((w.x | w.y | w.z | w.w) == 0u) continue;
                    const int j0 = c * 16;
                    const unsigned int words[4] = {w.x, w.y, w.z, w.w};
#pragma unroll
                    for (int u = 0; u < 4; ++u) {
#pragma unroll
                        for (int e = 0; e < 4; ++e) {
                            if ((words[u] >> (e * 8)) & 0xFFu) {
                                const int j = j0 + u * 4 + e;
                                const float v = sSp[j];
                                if (v > best) { best = v; bi = j; }
                            }
                        }
                    }
                }
                if (bi == i) sSel[i] += sPoss[i];
            }
        }
        __syncthreads();

        {
            const int k = 4 * t;
            unsigned int p = (sSel[k + 0] > 0.f ? 0x00000001u : 0u)
                           | (sSel[k + 1] > 0.f ? 0x00000100u : 0u)
                           | (sSel[k + 2] > 0.f ? 0x00010000u : 0u)
                           | (sSel[k + 3] > 0.f ? 0x01000000u : 0u);
            sSelW[t] = p;
        }
        __syncthreads();

        for (int tile = wave; tile < 64; tile += 8) {
            const int j0 = tile * 16;
            const uint8_t* brow = bmask + ((size_t)(j0 + (lane & 15)) << 10);
            v8i acc = {0, 0, 0, 0, 0, 0, 0, 0};
            for (int k0 = 0; k0 < N_BOXES; k0 += 64) {
                v8i A, B;
#pragma unroll
                for (int v = 0; v < 8; ++v) {
                    const int kstart = (v >> 1) * 16 + (v & 1) * 4 + half * 8;
                    A[v] = (int)sSelW[(k0 + kstart) >> 2];
                }
#pragma unroll
                for (int v = 0; v < 4; ++v) {
                    const int k = k0 + (v >> 1) * 16 + (v & 1) * 4 + half * 8;
                    B[v]     = *(const int*)(brow + k);
                    B[v + 4] = *(const int*)(brow + k + 32);
                }
                acc = __builtin_amdgcn_wmma_i32_16x16x64_iu8(
                    false, A, false, B, acc, false, false);
            }
            if (lane < 16) sBlocks[j0 + lane] = acc[0];
        }
        __syncthreads();

        for (int q = 0; q < 4; ++q) {
            const int j = t + q * 256;
            if (sBlocks[j] != 0) sPoss[j] = 0.f;
        }
        __syncthreads();
    }

    for (int q = 0; q < 4; ++q) {
        const int i = t + q * 256;
        out_mask[i * BATCH + b] = sSel[i];
    }
}

__global__ __launch_bounds__(32) void topk_kernel(
    const float* __restrict__ prob, const float* __restrict__ out_mask,
    int* __restrict__ topk_idx, int* __restrict__ batch_idx)
{
    const int b    = blockIdx.x;
    const int lane = threadIdx.x;

    float ms[32];
#pragma unroll
    for (int s = 0; s < 32; ++s) {
        const int g = lane * 32 + s;
        ms[s] = out_mask[g * BATCH + b] * fmaxf(prob[g * BATCH + b], 0.f);
    }

    for (int tK = 0; tK < N_OBJ; ++tK) {
        float bv = -1.f; int bidx = lane * 32;
#pragma unroll
        for (int s = 0; s < 32; ++s) {
            if (ms[s] > bv) { bv = ms[s]; bidx = lane * 32 + s; }
        }
#pragma unroll
        for (int off = 16; off > 0; off >>= 1) {
            const float ov = __shfl_xor(bv, off, 32);
            const int   oi = __shfl_xor(bidx, off, 32);
            if (ov > bv || (ov == bv && oi < bidx)) { bv = ov; bidx = oi; }
        }
        if (lane == 0) {
            topk_idx[tK * BATCH + b]  = bidx;
            batch_idx[tK * BATCH + b] = b;
        }
        if ((bidx >> 5) == lane) ms[bidx & 31] = -1.f;
    }
}

extern "C" void kernel_launch(void* const* d_in, const int* in_sizes, int n_in,
                              void* d_out, int out_size, void* d_ws, size_t ws_size,
                              hipStream_t stream) {
    const float* prob = (const float*)d_in[0];
    const float* bx   = (const float*)d_in[1];
    const float* by   = (const float*)d_in[2];
    const float* bw   = (const float*)d_in[3];
    const float* bh   = (const float*)d_in[4];

    uint8_t* mask = (uint8_t*)d_ws;
    float* out_mask = (float*)d_out;
    int* topk_idx  = (int*)((float*)d_out + N_BOXES * BATCH);
    int* batch_idx = topk_idx + N_OBJ * BATCH;

    build_mask_kernel<<<dim3(BATCH, 8), 256, 0, stream>>>(bx, by, bw, bh, mask);
    nms_select_kernel<<<BATCH, 256, 0, stream>>>(prob, mask, out_mask);
    topk_kernel<<<BATCH, 32, 0, stream>>>(prob, out_mask, topk_idx, batch_idx);
}
